// TransducerJoint_29429115912723
// MI455X (gfx1250) — matrix operand order repaired and checked by reading, not compiled, not run on hardware
//
#include <hip/hip_runtime.h>
#include <hip/hip_bf16.h>

// Problem constants (from reference setup_inputs)
#define BB 8
#define TT 200
#define UU 60
#define EE 512
#define PP 512
#define JJ 512
#define VV 4096

#define ROWPAD 260   // LDS row stride in u32 (256 data + 4 pad -> bank-conflict-free frag loads)

typedef __attribute__((ext_vector_type(16))) __bf16 v16bf;
typedef __attribute__((ext_vector_type(8)))  float  v8f;

__device__ __forceinline__ unsigned short f2bf(float f) {
    unsigned int u = __float_as_uint(f);
    u += 0x7FFFu + ((u >> 16) & 1u);
    return (unsigned short)(u >> 16);
}
__device__ __forceinline__ unsigned int pack2bf(float a, float b) {
    return (unsigned int)f2bf(a) | ((unsigned int)f2bf(b) << 16);
}

// Y[m, j] = dot(X[m, :K], W[j, :K]) + bias[j]   (N fixed = 512 = JJ)
__global__ __launch_bounds__(256) void proj_kernel(
    const float* __restrict__ X, const float* __restrict__ W,
    const float* __restrict__ bias, float* __restrict__ Y, int K)
{
    int idx = blockIdx.x * 256 + threadIdx.x;
    int j = idx & (JJ - 1);
    int m = idx >> 9;
    const float4* x4 = (const float4*)(X + (size_t)m * K);
    const float4* w4 = (const float4*)(W + (size_t)j * K);
    float acc = 0.0f;
    for (int k = 0; k < (K >> 2); ++k) {
        float4 a = x4[k];
        float4 b = w4[k];
        acc += a.x * b.x + a.y * b.y + a.z * b.z + a.w * b.w;
    }
    Y[idx] = acc + bias[j];
}

// fp32 -> packed bf16 pairs
__global__ __launch_bounds__(256) void cvt_bf16_kernel(
    const float* __restrict__ src, unsigned int* __restrict__ dst, int npairs)
{
    int i = blockIdx.x * 256 + threadIdx.x;
    if (i >= npairs) return;
    float2 v = ((const float2*)src)[i];
    dst[i] = pack2bf(v.x, v.y);
}

// Main joint GEMM.
// grid = (VV/1024, Mtotal/128), block = 256 (8 wave32).
// Each WG: stages 128x512 bf16 tanh(enc+pred) tile in LDS once, then loops over
// 8 N-groups of 128 cols; each wave owns 16 cols and 8 M-tiles (128 rows) so one
// B fragment feeds 8 WMMAs per K-step.
__global__ __launch_bounds__(256) void joint_gemm(
    const float* __restrict__ encp, const float* __restrict__ predp,
    const unsigned short* __restrict__ Wbf, const float* __restrict__ out_b,
    float* __restrict__ out)
{
    __shared__ unsigned int ldsA[128 * ROWPAD]; // ~130 KB (CDNA5 WGP has 320 KB LDS)

    const int tid = threadIdx.x;
    const int m0  = blockIdx.y * 128;

    // ---- Stage A tile: tanh(enc + pred) -> bf16 in LDS (once per WG) ----
    {
        int r  = tid >> 1;               // row 0..127
        int k0 = (tid & 1) * 256;        // half-row of 256 k-elements
        int m  = m0 + r;
        int b  = m / (TT * UU);
        int rem = m - b * (TT * UU);
        int t  = rem / UU;
        int u  = rem - t * UU;
        const float4* er = (const float4*)(encp  + ((size_t)(b * TT + t)) * JJ + k0);
        const float4* pr = (const float4*)(predp + ((size_t)(b * UU + u)) * JJ + k0);
        unsigned int* dst = ldsA + r * ROWPAD + (k0 >> 1);
#pragma unroll 4
        for (int i = 0; i < 64; ++i) {
            float4 e = er[i];
            float4 p = pr[i];
            dst[2 * i + 0] = pack2bf(tanhf(e.x + p.x), tanhf(e.y + p.y));
            dst[2 * i + 1] = pack2bf(tanhf(e.z + p.z), tanhf(e.w + p.w));
        }
    }
    __syncthreads();

    const int wave   = tid >> 5;
    const int lane   = tid & 31;
    const int nlane  = lane & 15;            // N column within tile / A row within tile
    const int khalfA = (lane >> 4) << 3;     // A frag: lanes 16-31 offset by 8 K-elems
    const int khalfB = (lane >> 4) << 3;     // B frag: lanes 16-31 offset by 8 K-elems

    for (int ng = 0; ng < 8; ++ng) {
        const int n = blockIdx.x * 1024 + ng * 128 + wave * 16 + nlane;

        v8f acc[8];
        {
            float bias = out_b[n];
#pragma unroll
            for (int mt = 0; mt < 8; ++mt)
#pragma unroll
                for (int i = 0; i < 8; ++i) acc[mt][i] = bias;
        }

        const unsigned short* wbase = Wbf + (size_t)n * JJ + khalfB;

#pragma unroll 4
        for (int kk = 0; kk < JJ; kk += 32) {
            // B fragment (32x16 bf16): K-elems khalfB..+7 and 16+khalfB..+7 per lane
            union { uint4 q[2]; v16bf v; } Bf;
            const uint4* wrow = (const uint4*)(wbase + kk);
            Bf.q[0] = wrow[0];
            Bf.q[1] = wrow[2];
#pragma unroll
            for (int mt = 0; mt < 8; ++mt) {
                const unsigned int* arow = ldsA + (mt * 16 + nlane) * ROWPAD;
                union { uint4 q[2]; v16bf v; } A;
                A.q[0] = *(const uint4*)(arow + ((kk + khalfA) >> 1));
                A.q[1] = *(const uint4*)(arow + ((kk + 16 + khalfA) >> 1));
                acc[mt] = __builtin_amdgcn_wmma_f32_16x16x32_bf16(
                    false, A.v, false, Bf.v, (short)0, acc[mt], false, false);
            }
        }

        // ---- Store 128x16 fp32, non-temporal (stream past L2, keep Wbf hot) ----
#pragma unroll
        for (int mt = 0; mt < 8; ++mt) {
#pragma unroll
            for (int i = 0; i < 8; ++i) {
                int r2 = (lane < 16) ? i : (i + 8);
                __builtin_nontemporal_store(
                    acc[mt][i], &out[((size_t)(m0 + mt * 16 + r2)) * VV + n]);
            }
        }
    }
}

extern "C" void kernel_launch(void* const* d_in, const int* in_sizes, int n_in,
                              void* d_out, int out_size, void* d_ws, size_t ws_size,
                              hipStream_t stream) {
    const float* enc_out  = (const float*)d_in[0]; // [B,T,E]
    const float* pred_out = (const float*)d_in[1]; // [B,U,P]
    const float* enc_W    = (const float*)d_in[2]; // [J,E]
    const float* enc_b    = (const float*)d_in[3]; // [J]
    const float* pred_W   = (const float*)d_in[4]; // [J,P]
    const float* pred_b   = (const float*)d_in[5]; // [J]
    const float* out_W    = (const float*)d_in[6]; // [V,J]
    const float* out_b    = (const float*)d_in[7]; // [V]
    float* out = (float*)d_out;

    char* ws = (char*)d_ws;
    float* encp          = (float*)ws;                          // B*T*J fp32 = 3,276,800 B
    float* predp         = (float*)(ws + 3276800);              // B*U*J fp32 =   983,040 B
    unsigned short* Wbf  = (unsigned short*)(ws + 4259840);     // V*J bf16  = 4,194,304 B

    // 1) prejoin projections (fp32, bias fused)
    proj_kernel<<<(BB * TT * JJ) / 256, 256, 0, stream>>>(enc_out, enc_W, enc_b, encp, EE);
    proj_kernel<<<(BB * UU * JJ) / 256, 256, 0, stream>>>(pred_out, pred_W, pred_b, predp, PP);

    // 2) out_W -> bf16 (L2-resident operand for WMMA)
    cvt_bf16_kernel<<<((VV * JJ) / 2) / 256, 256, 0, stream>>>(out_W, (unsigned int*)Wbf, (VV * JJ) / 2);

    // 3) main joint + vocab GEMM via v_wmma_f32_16x16x32_bf16
    dim3 grid(VV / 1024, (BB * TT * UU) / 128);
    joint_gemm<<<grid, 256, 0, stream>>>(encp, predp, Wbf, out_b, out);
}
